// SymmetricContraction_7164005450047
// MI455X (gfx1250) — matrix operand order repaired and checked by reading, not compiled, not run on hardware
//
#include <hip/hip_runtime.h>
#include <hip/hip_bf16.h>

// SymmetricContraction (MACE-style) for MI455X / gfx1250.
// T3[p,ki] = X3[p,abj] * U3[abj,ki] as bf16 WMMA GEMM (f32 accumulate).
// B (packed bf16 u3) is streamed into LDS with DOUBLE-BUFFERED async
// global->LDS copies (ASYNCcnt), overlapping with the 128 WMMAs of the
// current K-block. A-fragments are built in registers right before use.

typedef __attribute__((ext_vector_type(16))) __bf16 v16bf;
typedef __attribute__((ext_vector_type(8)))  float  v8f;

#define NATOMS 512
#define CCH    128
#define DD     16
#define NPAIR  (NATOMS * CCH)     // 65536
#define TCOLS  176                // 128 (T3) + 32 (T2) + 16 (T1)
#define B3_STRIDE 1040            // 512 abj * 2B + 16B pad (bank spread)
#define B2_STRIDE 528             // 256*2 + 16
#define B1_STRIDE 80              // 32*2  + 16
#define B3_BUF    133120          // 128 cols * B3_STRIDE
#define XSTASH    (2 * B3_BUF)    // 266240: per-wave x stash (16 rows * 64B)
#define SMEM_BYTES (XSTASH + 8 * 1024)   // 274432

union FragBF {
  v16bf    v;
  unsigned u[8];
  uint4    q[2];
};

__device__ __forceinline__ unsigned pk_bf16(float lo, float hi) {
  union { __hip_bfloat16 h[2]; unsigned u; } t;
  t.h[0] = __float2bfloat16(lo);
  t.h[1] = __float2bfloat16(hi);
  return t.u;
}

__device__ __forceinline__ v16bf make_afrag(float pa0, float pa1, const float xjf[8]) {
  FragBF af;
#pragma unroll
  for (int m = 0; m < 4; ++m) af.u[m]     = pk_bf16(pa0 * xjf[2*m], pa0 * xjf[2*m+1]);
#pragma unroll
  for (int m = 0; m < 4; ++m) af.u[4 + m] = pk_bf16(pa1 * xjf[2*m], pa1 * xjf[2*m+1]);
  return af.v;
}

__device__ __forceinline__ v16bf load_bfrag(const unsigned char* bp) {
  FragBF bf;
  bf.q[0] = *(const uint4*)(bp);
  bf.q[1] = *(const uint4*)(bp + 32);
  return bf.v;
}

__device__ __forceinline__ void async_wait_all() {
#if __has_builtin(__builtin_amdgcn_s_wait_asynccnt)
  __builtin_amdgcn_s_wait_asynccnt(0);
#else
  asm volatile("s_wait_asynccnt 0x0" ::: "memory");
#endif
}

// Async-stage one 128-col x 512-abj B3 slab (kb-th K-block) into LDS buffer
// at LDS byte address `dstBase`. Each thread issues 32 b128 async copies.
__device__ __forceinline__ void stage_b3_async(const unsigned char* __restrict__ Bu3,
                                               unsigned dstBase, int tid, int kb) {
  const unsigned ldsOff0 = dstBase + (unsigned)((tid >> 6) * B3_STRIDE + (tid & 63) * 16);
  const unsigned gOff0   = (unsigned)((tid >> 6) * 8192 + kb * 1024 + (tid & 63) * 16);
#pragma unroll
  for (int k = 0; k < 32; ++k) {
    // GVS mode: mem = SGPR_base + VGPR_off;  LDS dst = LDS_BASE + VGPR (dsaddr)
    asm volatile("global_load_async_to_lds_b128 %0, %1, %2"
                 :
                 : "v"(ldsOff0 + (unsigned)(k * 4160)),   // +4*B3_STRIDE per step
                   "v"(gOff0 + (unsigned)(k * 32768)),    // +4*8192B per step
                   "s"(Bu3)
                 : "memory");
  }
}

// ---------------- prep kernels: pack u tensors into bf16 B matrices ----------
// Bu3 packed position (col,pos): pos = (kb*16+ch)*32 + bl*16 + j holds
// u3[a=ch][b=2*kb+bl][j] for column col (k,i slot).
__global__ __launch_bounds__(256) void pack_u3_k(const float* __restrict__ u3_0e,
                                                 const float* __restrict__ u3_1o,
                                                 unsigned short* __restrict__ Bu3) {
  int g = blockIdx.x * 256 + threadIdx.x;   // 0 .. 128*4096-1
  int col = g >> 12, pos = g & 4095;
  int gc = pos >> 5, K = pos & 31;
  int kb = gc >> 4, ch = gc & 15;
  int bl = K >> 4, j = K & 15;
  int abj = ch * 256 + (2 * kb + bl) * 16 + j;   // original a*256+b*16+j
  float v = 0.f;
  if (col < 23)       v = u3_0e[abj * 23 + col];
  else if (col < 122) v = u3_1o[abj * 99 + (col - 23)];
  union { __hip_bfloat16 h; unsigned short s; } c;
  c.h = __float2bfloat16(v);
  Bu3[col * 4096 + pos] = c.s;
}

__global__ __launch_bounds__(256) void pack_u2_k(const float* __restrict__ u2_0e,
                                                 const float* __restrict__ u2_1o,
                                                 unsigned short* __restrict__ Bu2) {
  int g = blockIdx.x * 256 + threadIdx.x;   // 0 .. 8191
  int col = g >> 8, ab = g & 255;
  float v = 0.f;
  if (col < 4)       v = u2_0e[ab * 4 + col];
  else if (col < 22) v = u2_1o[ab * 18 + (col - 4)];
  union { __hip_bfloat16 h; unsigned short s; } c;
  c.h = __float2bfloat16(v);
  Bu2[col * 256 + ab] = c.s;
}

__global__ __launch_bounds__(256) void pack_u1_k(const float* __restrict__ u1_0e,
                                                 const float* __restrict__ u1_1o,
                                                 unsigned short* __restrict__ Bu1) {
  int g = blockIdx.x * 256 + threadIdx.x;   // 0 .. 511
  int col = g >> 5, k = g & 31;
  float v = 0.f;
  if (k < 16) {
    if (col == 0)     v = u1_0e[k];
    else if (col < 4) v = u1_1o[k * 3 + (col - 1)];
  }
  union { __hip_bfloat16 h; unsigned short s; } c;
  c.h = __float2bfloat16(v);
  Bu1[col * 32 + k] = c.s;
}

// ---------------- main WMMA kernel ----------------
__global__ __launch_bounds__(256) void sc_main_k(const float* __restrict__ x,
                                                 const unsigned char* __restrict__ Bu3,
                                                 const unsigned char* __restrict__ Bu2,
                                                 const unsigned char* __restrict__ Bu1,
                                                 float* __restrict__ T) {
  extern __shared__ unsigned char sm[];
  const unsigned smaddr = (unsigned)(size_t)sm;   // LDS byte address of sm[0]
  const int tid  = threadIdx.x;
  const int wave = tid >> 5;
  const int lane = tid & 31;
  const int c16  = lane & 15;
  const int hi   = lane >> 4;
  const int pt   = blockIdx.x * 8 + wave;        // pair-tile (16 pairs)
  const int prow = pt * 16 + c16;                // this lane's pair row

  // load this row's x (16 f32); lanes l and l+16 hold the same row
  float xr[16];
  {
    const float4* xp = (const float4*)(x + (size_t)prow * DD);
#pragma unroll
    for (int m = 0; m < 4; ++m) {
      float4 v = xp[m];
      xr[4*m+0] = v.x; xr[4*m+1] = v.y; xr[4*m+2] = v.z; xr[4*m+3] = v.w;
    }
  }
  float xjf[8];
#pragma unroll
  for (int m = 0; m < 8; ++m) xjf[m] = hi ? xr[8 + m] : xr[m];

  const int bvoff = hi * 32;   // x stash: hi half stores x[8..15] (+32B)

  // kick off async staging of K-block 0 into buffer 0
  stage_b3_async(Bu3, smaddr, tid, 0);

  // stash this wave's x rows in LDS for dynamic-index access (b values)
  const int xbase = XSTASH + wave * 1024 + c16 * 64;
  {
    float4 s0 = {xjf[0], xjf[1], xjf[2], xjf[3]};
    float4 s1 = {xjf[4], xjf[5], xjf[6], xjf[7]};
    *(float4*)(sm + xbase + bvoff)      = s0;
    *(float4*)(sm + xbase + bvoff + 16) = s1;
  }

  v8f acc[8];
#pragma unroll
  for (int nt = 0; nt < 8; ++nt)
    acc[nt] = (v8f){0.f, 0.f, 0.f, 0.f, 0.f, 0.f, 0.f, 0.f};

  async_wait_all();
  __syncthreads();

#pragma unroll 1
  for (int kb = 0; kb < 8; ++kb) {
    // prefetch next K-block into the other buffer while we compute
    if (kb + 1 < 8)
      stage_b3_async(Bu3, smaddr + (((kb + 1) & 1) ? (unsigned)B3_BUF : 0u),
                     tid, kb + 1);

    const unsigned char* buf = sm + ((kb & 1) ? B3_BUF : 0);

    // x_b pair for this kb (dynamic index via LDS stash)
    const float2 xb = *(const float2*)(sm + xbase + kb * 8);

#pragma unroll
    for (int ch = 0; ch < 16; ++ch) {     // a = ch (compile-time xr index)
      const float pa0 = xr[ch] * xb.x;
      const float pa1 = xr[ch] * xb.y;
      v16bf af = make_afrag(pa0, pa1, xjf);
#pragma unroll
      for (int nt = 0; nt < 8; ++nt) {
        v16bf bf = load_bfrag(buf + (nt * 16 + c16) * B3_STRIDE + ch * 64 + hi * 16);
        acc[nt] = __builtin_amdgcn_wmma_f32_16x16x32_bf16(false, af, false, bf,
                                                          (short)0, acc[nt], false, false);
      }
    }
    // my async copies for kb+1 done; barrier => everyone's done reading/writing
    async_wait_all();
    __syncthreads();
  }
#pragma unroll
  for (int nt = 0; nt < 8; ++nt)
#pragma unroll
    for (int v = 0; v < 8; ++v)
      T[(size_t)(pt * 16 + v + hi * 8) * TCOLS + nt * 16 + c16] = acc[nt][v];

  // ---- T2: 2 ntiles, K = 256 over ab (whole B2 staged at once, buffer 0) ----
  {
#pragma unroll
    for (int k = 0; k < 4; ++k) {
      int idx = tid + k * 256;               // 0..1023 chunks of 16B
      int col = idx >> 5, seg = idx & 31;
      *(uint4*)(sm + col * B2_STRIDE + seg * 16) =
          *(const uint4*)(Bu2 + ((size_t)col * 256 + seg * 8) * 2);
    }
    __syncthreads();
    v8f acc2[2];
    acc2[0] = (v8f){0.f, 0.f, 0.f, 0.f, 0.f, 0.f, 0.f, 0.f};
    acc2[1] = (v8f){0.f, 0.f, 0.f, 0.f, 0.f, 0.f, 0.f, 0.f};
#pragma unroll
    for (int ch = 0; ch < 8; ++ch) {         // ab base = ch*32, a in {2ch,2ch+1}
      const float pa0 = xr[2 * ch];
      const float pa1 = xr[2 * ch + 1];
      v16bf af = make_afrag(pa0, pa1, xjf);
#pragma unroll
      for (int nt2 = 0; nt2 < 2; ++nt2) {
        v16bf bf = load_bfrag(sm + (nt2 * 16 + c16) * B2_STRIDE + ch * 64 + hi * 16);
        acc2[nt2] = __builtin_amdgcn_wmma_f32_16x16x32_bf16(false, af, false, bf,
                                                            (short)0, acc2[nt2], false, false);
      }
    }
#pragma unroll
    for (int nt2 = 0; nt2 < 2; ++nt2)
#pragma unroll
      for (int v = 0; v < 8; ++v)
        T[(size_t)(pt * 16 + v + hi * 8) * TCOLS + 128 + nt2 * 16 + c16] = acc2[nt2][v];
    __syncthreads();
  }

  // ---- T1: 1 tile, K = 16 (padded to 32) ----
  {
    if (tid < 64) {
      int col = tid >> 2, seg = tid & 3;
      *(uint4*)(sm + col * B1_STRIDE + seg * 16) =
          *(const uint4*)(Bu1 + ((size_t)col * 32 + seg * 8) * 2);
    }
    __syncthreads();
    v8f acc1 = {0.f, 0.f, 0.f, 0.f, 0.f, 0.f, 0.f, 0.f};
    FragBF af;
#pragma unroll
    for (int m = 0; m < 4; ++m) af.u[m] = pk_bf16(xjf[2*m], xjf[2*m+1]);
#pragma unroll
    for (int m = 0; m < 4; ++m) af.u[4 + m] = 0u;
    v16bf bf = load_bfrag(sm + c16 * B1_STRIDE + hi * 16);
    acc1 = __builtin_amdgcn_wmma_f32_16x16x32_bf16(false, af.v, false, bf,
                                                   (short)0, acc1, false, false);
#pragma unroll
    for (int v = 0; v < 8; ++v)
      T[(size_t)(pt * 16 + v + hi * 8) * TCOLS + 160 + c16] = acc1[v];
  }
}

// ---------------- combine kernel: species-weight reduction ----------------
__global__ __launch_bounds__(256) void sc_combine_k(const float* __restrict__ T,
                                                    const int* __restrict__ index,
                                                    const float* __restrict__ w3_0e,
                                                    const float* __restrict__ w2_0e,
                                                    const float* __restrict__ w1_0e,
                                                    const float* __restrict__ w3_1o,
                                                    const float* __restrict__ w2_1o,
                                                    const float* __restrict__ w1_1o,
                                                    float* __restrict__ out) {
  const int p = blockIdx.x * 256 + threadIdx.x;    // pair id
  const int n = p >> 7, c = p & 127;
  const int s = index[n];
  const float* t = T + (size_t)p * TCOLS;

  float a0 = 0.f;
#pragma unroll
  for (int k = 0; k < 23; ++k) a0 += w3_0e[(s * 23 + k) * 128 + c] * t[k];
#pragma unroll
  for (int k = 0; k < 4; ++k)  a0 += w2_0e[(s * 4 + k) * 128 + c] * t[128 + k];
  a0 += w1_0e[s * 128 + c] * t[160];

  float a1 = 0.f, a2 = 0.f, a3 = 0.f;
#pragma unroll
  for (int k = 0; k < 33; ++k) {
    const float w = w3_1o[(s * 33 + k) * 128 + c];
    a1 += w * t[23 + 3 * k];
    a2 += w * t[24 + 3 * k];
    a3 += w * t[25 + 3 * k];
  }
#pragma unroll
  for (int k = 0; k < 6; ++k) {
    const float w = w2_1o[(s * 6 + k) * 128 + c];
    a1 += w * t[132 + 3 * k];
    a2 += w * t[133 + 3 * k];
    a3 += w * t[134 + 3 * k];
  }
  {
    const float w = w1_1o[s * 128 + c];
    a1 += w * t[161]; a2 += w * t[162]; a3 += w * t[163];
  }
  float4 o = {a0, a1, a2, a3};
  *(float4*)(out + (size_t)p * 4) = o;
}

extern "C" void kernel_launch(void* const* d_in, const int* in_sizes, int n_in,
                              void* d_out, int out_size, void* d_ws, size_t ws_size,
                              hipStream_t stream) {
  (void)in_sizes; (void)n_in; (void)out_size; (void)ws_size;
  const float* x     = (const float*)d_in[0];
  const int*   index = (const int*)  d_in[1];
  const float* u3_0e = (const float*)d_in[2];
  const float* w3_0e = (const float*)d_in[3];
  const float* u2_0e = (const float*)d_in[4];
  const float* w2_0e = (const float*)d_in[5];
  const float* u1_0e = (const float*)d_in[6];
  const float* w1_0e = (const float*)d_in[7];
  const float* u3_1o = (const float*)d_in[8];
  const float* w3_1o = (const float*)d_in[9];
  const float* u2_1o = (const float*)d_in[10];
  const float* w2_1o = (const float*)d_in[11];
  const float* u1_1o = (const float*)d_in[12];
  const float* w1_1o = (const float*)d_in[13];

  unsigned char* ws = (unsigned char*)d_ws;
  unsigned short* Bu3 = (unsigned short*)(ws);                      // 1 MB
  unsigned short* Bu2 = (unsigned short*)(ws + (1u << 20));         // 16 KB
  unsigned short* Bu1 = (unsigned short*)(ws + (1u << 20) + 16384); // 1 KB
  float*          T   = (float*)(ws + (2u << 20));                  // 65536*176*4 B

  pack_u3_k<<<2048, 256, 0, stream>>>(u3_0e, u3_1o, Bu3);
  pack_u2_k<<<32,   256, 0, stream>>>(u2_0e, u2_1o, Bu2);
  pack_u1_k<<<2,    256, 0, stream>>>(u1_0e, u1_1o, Bu1);

  sc_main_k<<<512, 256, SMEM_BYTES, stream>>>(x,
      (const unsigned char*)Bu3, (const unsigned char*)Bu2,
      (const unsigned char*)Bu1, T);

  sc_combine_k<<<256, 256, 0, stream>>>(T, index, w3_0e, w2_0e, w1_0e,
                                        w3_1o, w2_1o, w1_1o, (float*)d_out);
}
